// Transformer_88905823027631
// MI455X (gfx1250) — matrix operand order repaired and checked by reading, not compiled, not run on hardware
//
#include <hip/hip_runtime.h>
#include <hip/hip_bf16.h>
#include <math.h>

// ---------------------------------------------------------------------------
// CDNA5 (gfx1250) transformer forward.
// GEMMs -> bf16 WMMA (v_wmma_f32_16x16x32_bf16), f32 accumulate.
// Attention K/V staged with global_load_async_to_lds_b128 (ASYNCcnt).
// f32->bf16 via vector fptrunc -> packed v_cvt_pk_bf16_f32.
// ---------------------------------------------------------------------------

typedef __attribute__((ext_vector_type(16))) __bf16    v16bf;
typedef __attribute__((ext_vector_type(2)))  __bf16    v2bf;
typedef __attribute__((ext_vector_type(8)))  float     v8f;
typedef __attribute__((ext_vector_type(4)))  float     v4f;
typedef __attribute__((ext_vector_type(2)))  float     v2f;
typedef __attribute__((ext_vector_type(4)))  unsigned  v4u;

#define TNL 6
#define TD  512
#define THD 8
#define TDK 64
#define TV  32000
#define TBB 8
#define TS  128
#define TT  128
#define TFF 2048
#define TPAD 1

#define BM 64
#define BN 128
#define BK 32
#define GFLAG_RELU   1
#define GFLAG_BTRANS 2

// pack two f32 -> one dword of two bf16: vector fptrunc selects a single
// v_cvt_pk_bf16_f32 with both sources.
__device__ __forceinline__ unsigned bf2pack(float lo, float hi) {
  v2f f; f[0] = lo; f[1] = hi;
  v2bf r = __builtin_convertvector(f, v2bf);
  unsigned u;
  __builtin_memcpy(&u, &r, 4);
  return u;
}

// ---------------------------------------------------------------------------
// GEMM: C[M,N] = epilogue(A[M,K] @ B[K,N] + bias), epilogue = relu / +residual
// flags bit1: B stored transposed as Bt[N,K] (generator: tok_emb^T).
// REQUIRES: M%64==0, N%128==0, K%32==0 (true for every call in this model).
// Block tile 64x128, BK=32 (one WMMA K-step). 8 waves, each 32x32 = 2x2 WMMAs.
// Software-pipelined: regs hold tile k+1 while LDS tile k feeds the WMMAs.
// ---------------------------------------------------------------------------
__global__ __launch_bounds__(256) void gemm_bf16_wmma(
    const float* __restrict__ A, const float* __restrict__ B,
    const float* __restrict__ bias, const float* __restrict__ resid,
    float* __restrict__ C, int M, int N, int K, int flags)
{
  __shared__ __bf16 As[BM][BK + 8];   // 80B row stride -> 16B-aligned chunks
  __shared__ __bf16 Bs[BN][BK + 8];   // [N][K] layout, K-contiguous

  const int tid  = threadIdx.x;
  const int lane = tid & 31;
  const int wave = tid >> 5;     // 0..7
  const int wm   = wave >> 2;    // 0..1  (M)
  const int wn   = wave & 3;     // 0..3  (N)
  const int row0 = blockIdx.y * BM;
  const int col0 = blockIdx.x * BN;
  const bool btrans = (flags & GFLAG_BTRANS) != 0;

  v8f acc[2][2];
  #pragma unroll
  for (int im = 0; im < 2; ++im)
    #pragma unroll
    for (int jn = 0; jn < 2; ++jn)
      #pragma unroll
      for (int vv = 0; vv < 8; ++vv) acc[im][jn][vv] = 0.0f;

  // staging coordinates
  const int arow  = tid >> 2;          // 0..63, 8 consecutive K per thread
  const int acol  = (tid & 3) * 8;     // 0/8/16/24
  const int bn_n  = tid & 127;         // normal-B: column, coalesced across tid
  const int bn_k0 = (tid >> 7) * 16;   // 16 K values per thread
  const int bt_n  = tid >> 1;          // transposed-B: row
  const int bt_k0 = (tid & 1) * 16;

  v4f areg[2];
  v4f breg[4];

  auto fetchA = [&](int kk) {
    const float* p = A + (size_t)(row0 + arow) * K + kk + acol;
    areg[0] = *(const v4f*)p;
    areg[1] = *(const v4f*)(p + 4);
  };
  auto fetchB = [&](int kk) {
    if (btrans) {
      const float* p = B + (size_t)(col0 + bt_n) * K + kk + bt_k0;
      breg[0] = *(const v4f*)(p + 0);
      breg[1] = *(const v4f*)(p + 4);
      breg[2] = *(const v4f*)(p + 8);
      breg[3] = *(const v4f*)(p + 12);
    } else {
      const float* p = B + (size_t)(kk + bn_k0) * N + (col0 + bn_n);
      #pragma unroll
      for (int j = 0; j < 16; ++j) breg[j >> 2][j & 3] = p[(size_t)j * N];
    }
  };
  auto stageLDS = [&]() {
    v4u pa;
    pa[0] = bf2pack(areg[0][0], areg[0][1]);
    pa[1] = bf2pack(areg[0][2], areg[0][3]);
    pa[2] = bf2pack(areg[1][0], areg[1][1]);
    pa[3] = bf2pack(areg[1][2], areg[1][3]);
    *(v4u*)&As[arow][acol] = pa;

    v4u p0, p1;
    p0[0] = bf2pack(breg[0][0], breg[0][1]);
    p0[1] = bf2pack(breg[0][2], breg[0][3]);
    p0[2] = bf2pack(breg[1][0], breg[1][1]);
    p0[3] = bf2pack(breg[1][2], breg[1][3]);
    p1[0] = bf2pack(breg[2][0], breg[2][1]);
    p1[1] = bf2pack(breg[2][2], breg[2][3]);
    p1[2] = bf2pack(breg[3][0], breg[3][1]);
    p1[3] = bf2pack(breg[3][2], breg[3][3]);
    const int n   = btrans ? bt_n : bn_n;
    const int kk0 = btrans ? bt_k0 : bn_k0;
    *(v4u*)&Bs[n][kk0 + 0] = p0;
    *(v4u*)&Bs[n][kk0 + 8] = p1;
  };

  fetchA(0);
  fetchB(0);

  for (int k0 = 0; k0 < K; k0 += BK) {
    stageLDS();
    __syncthreads();

    if (k0 + BK < K) {          // uniform branch: prefetch next tile into regs
      __builtin_prefetch(&A[(size_t)(row0 + arow) * K + k0 + 2 * BK], 0, 1);
      fetchA(k0 + BK);
      fetchB(k0 + BK);
    }

    // ---- fragments per ISA 7.12.2 layout ----
    v16bf afrag[2], bfrag[2];
    #pragma unroll
    for (int im = 0; im < 2; ++im) {
      const int m  = wm * 32 + im * 16 + (lane & 15);
      const int kb = (lane & 16) ? 8 : 0;  // lanes 0-15: K 0-7/16-23; 16-31: 8-15/24-31
      #pragma unroll
      for (int j = 0; j < 8; ++j) {
        afrag[im][j]     = As[m][kb + j];
        afrag[im][8 + j] = As[m][kb + 16 + j];
      }
    }
    #pragma unroll
    for (int jn = 0; jn < 2; ++jn) {
      const int n  = wn * 32 + jn * 16 + (lane & 15);
      const int kb = (lane & 16) ? 8 : 0;  // lanes 0-15: K 0-7/16-23; 16-31: 8-15/24-31
      #pragma unroll
      for (int j = 0; j < 8; ++j) {
        bfrag[jn][j]     = Bs[n][kb + j];
        bfrag[jn][8 + j] = Bs[n][kb + 16 + j];
      }
    }

    #pragma unroll
    for (int im = 0; im < 2; ++im)
      #pragma unroll
      for (int jn = 0; jn < 2; ++jn)
        acc[im][jn] = __builtin_amdgcn_wmma_f32_16x16x32_bf16(
            false, afrag[im], false, bfrag[jn],
            (short)0, acc[im][jn], false, false);
    __syncthreads();
  }

  // ---- epilogue (exact tiles, no guards): bias / relu / residual ----
  const bool relu = (flags & GFLAG_RELU) != 0;
  #pragma unroll
  for (int im = 0; im < 2; ++im) {
    #pragma unroll
    for (int jn = 0; jn < 2; ++jn) {
      const int mbase = row0 + wm * 32 + im * 16 + ((lane >> 4) * 8);
      const int ncol  = col0 + wn * 32 + jn * 16 + (lane & 15);
      #pragma unroll
      for (int vv = 0; vv < 8; ++vv) {
        float r = acc[im][jn][vv];
        if (bias)  r += bias[ncol];
        if (relu)  r = fmaxf(r, 0.0f);
        const size_t idx = (size_t)(mbase + vv) * N + ncol;
        if (resid) r += resid[idx];
        C[idx] = r;
      }
    }
  }
}

// ---------------------------------------------------------------------------
// Attention: one block per (head, batch). K/V staged into LDS with
// global_load_async_to_lds_b128 (per-lane async DMA, drained by asynccnt),
// then 2-pass online softmax per query row.
// ---------------------------------------------------------------------------
__global__ __launch_bounds__(128) void attention_kernel(
    const float* __restrict__ Q, const float* __restrict__ K,
    const float* __restrict__ V, float* __restrict__ O,
    const int* __restrict__ ktoks, int causal, int Tq, int Tk)
{
  __shared__ float Ks[TS * TDK];
  __shared__ float Vs[TS * TDK];
  __shared__ int   mk[TS];

  const int h = blockIdx.x, b = blockIdx.y;
  const int t = threadIdx.x;

  // ---- async stage: thread t DMAs key/value row t (64 floats = 16 B128s) ----
  {
    const float* kg = K + ((size_t)(b * Tk + t) * TD + h * TDK);
    const float* vg = V + ((size_t)(b * Tk + t) * TD + h * TDK);
    const unsigned kl = (unsigned)(uintptr_t)&Ks[t * TDK];  // LDS byte offset
    const unsigned vl = (unsigned)(uintptr_t)&Vs[t * TDK];
    #pragma unroll
    for (int c = 0; c < TDK; c += 4) {
      asm volatile("global_load_async_to_lds_b128 %0, %1, off"
                   :: "v"(kl + c * 4), "v"(kg + c) : "memory");
      asm volatile("global_load_async_to_lds_b128 %0, %1, off"
                   :: "v"(vl + c * 4), "v"(vg + c) : "memory");
    }
    mk[t] = (ktoks[b * Tk + t] == TPAD) ? 1 : 0;
    asm volatile("s_wait_asynccnt 0x0" ::: "memory");
  }
  __syncthreads();

  float qreg[TDK];
  const float* qp = Q + ((size_t)(b * Tq + t) * TD + h * TDK);
  #pragma unroll 8
  for (int d = 0; d < TDK; ++d) qreg[d] = qp[d];

  const float scale = 0.125f;   // 1/sqrt(64)
  float m = -3.0e38f;
  for (int s = 0; s < Tk; ++s) {
    if (mk[s] || (causal && s > t)) continue;
    float dot = 0.f;
    #pragma unroll 8
    for (int d = 0; d < TDK; ++d) dot += qreg[d] * Ks[s * TDK + d];
    m = fmaxf(m, dot * scale);
  }

  float sum = 0.f;
  float accv[TDK];
  #pragma unroll 8
  for (int d = 0; d < TDK; ++d) accv[d] = 0.f;
  for (int s = 0; s < Tk; ++s) {
    if (mk[s] || (causal && s > t)) continue;
    float dot = 0.f;
    #pragma unroll 8
    for (int d = 0; d < TDK; ++d) dot += qreg[d] * Ks[s * TDK + d];
    const float e = __expf(dot * scale - m);
    sum += e;
    #pragma unroll 8
    for (int d = 0; d < TDK; ++d) accv[d] += e * Vs[s * TDK + d];
  }
  const float inv = (sum > 0.f) ? (1.0f / sum) : 0.f;
  float* op = O + ((size_t)(b * Tq + t) * TD + h * TDK);
  #pragma unroll 8
  for (int d = 0; d < TDK; ++d) op[d] = accv[d] * inv;
}

// ---------------------------------------------------------------------------
// LayerNorm over D=512, one row per block (128 threads x 4 elems).
// ---------------------------------------------------------------------------
__global__ __launch_bounds__(128) void layernorm_kernel(
    const float* __restrict__ x, const float* __restrict__ g,
    const float* __restrict__ bta, float* __restrict__ out)
{
  __shared__ float s1[128], s2[128];
  const int row = blockIdx.x, tid = threadIdx.x;
  const float* xr = x + (size_t)row * TD;
  float v0[4]; float a = 0.f, sq = 0.f;
  #pragma unroll
  for (int i = 0; i < 4; ++i) {
    const float f = xr[tid + i * 128];
    v0[i] = f; a += f; sq += f * f;
  }
  s1[tid] = a; s2[tid] = sq;
  __syncthreads();
  for (int off = 64; off > 0; off >>= 1) {
    if (tid < off) { s1[tid] += s1[tid + off]; s2[tid] += s2[tid + off]; }
    __syncthreads();
  }
  const float mean = s1[0] * (1.0f / TD);
  const float var  = s2[0] * (1.0f / TD) - mean * mean;
  const float rstd = rsqrtf(var + 1e-6f);
  #pragma unroll
  for (int i = 0; i < 4; ++i) {
    const int d = tid + i * 128;
    out[(size_t)row * TD + d] = (v0[i] - mean) * rstd * g[d] + bta[d];
  }
}

// ---------------------------------------------------------------------------
// Token + positional embedding.
// ---------------------------------------------------------------------------
__global__ __launch_bounds__(256) void embed_kernel(
    const int* __restrict__ toks, const float* __restrict__ temb,
    const float* __restrict__ pemb, float* __restrict__ out, int total)
{
  const int idx = blockIdx.x * blockDim.x + threadIdx.x;
  if (idx >= total) return;
  const int row = idx / TD;
  const int d   = idx % TD;
  const int t   = row % TT;
  const int tok = toks[row];
  out[idx] = temb[(size_t)tok * TD + d] + pemb[(size_t)t * TD + d];
}

// ---------------------------------------------------------------------------
// In-place log_softmax over V=32000, one row per block.
// ---------------------------------------------------------------------------
__global__ __launch_bounds__(256) void log_softmax_kernel(float* __restrict__ logits)
{
  __shared__ float red[256];
  const int row = blockIdx.x, tid = threadIdx.x;
  float* lr = logits + (size_t)row * TV;

  float m = -3.0e38f;
  for (int i = tid; i < TV; i += 256) m = fmaxf(m, lr[i]);
  red[tid] = m; __syncthreads();
  for (int off = 128; off > 0; off >>= 1) {
    if (tid < off) red[tid] = fmaxf(red[tid], red[tid + off]);
    __syncthreads();
  }
  m = red[0]; __syncthreads();

  float sum = 0.f;
  for (int i = tid; i < TV; i += 256) sum += __expf(lr[i] - m);
  red[tid] = sum; __syncthreads();
  for (int off = 128; off > 0; off >>= 1) {
    if (tid < off) red[tid] += red[tid + off];
    __syncthreads();
  }
  const float lse = m + __logf(red[0]);
  for (int i = tid; i < TV; i += 256) lr[i] = lr[i] - lse;
}

// ---------------------------------------------------------------------------
// Host orchestration
// ---------------------------------------------------------------------------
static inline void gemm(hipStream_t s, const float* A, const float* B,
                        const float* bias, const float* resid, float* C,
                        int M, int N, int K, int flags) {
  dim3 g((N + BN - 1) / BN, (M + BM - 1) / BM);
  gemm_bf16_wmma<<<g, 256, 0, s>>>(A, B, bias, resid, C, M, N, K, flags);
}

extern "C" void kernel_launch(void* const* d_in, const int* in_sizes, int n_in,
                              void* d_out, int out_size, void* d_ws, size_t ws_size,
                              hipStream_t stream) {
  (void)in_sizes; (void)n_in; (void)out_size; (void)ws_size;

  const int*   src        = (const int*)d_in[0];
  const int*   tgt        = (const int*)d_in[1];
  const float* tok_emb    = (const float*)d_in[2];
  const float* pos_emb    = (const float*)d_in[3];
  const float* enc_attn_w = (const float*)d_in[4];
  const float* enc_attn_b = (const float*)d_in[5];
  const float* enc_ffn_w1 = (const float*)d_in[6];
  const float* enc_ffn_b1 = (const float*)d_in[7];
  const float* enc_ffn_w2 = (const float*)d_in[8];
  const float* enc_ffn_b2 = (const float*)d_in[9];
  const float* enc_ln     = (const float*)d_in[10];
  const float* enc_fln    = (const float*)d_in[11];
  const float* dec_attn_w = (const float*)d_in[12];
  const float* dec_attn_b = (const float*)d_in[13];
  const float* dec_ffn_w1 = (const float*)d_in[14];
  const float* dec_ffn_b1 = (const float*)d_in[15];
  const float* dec_ffn_w2 = (const float*)d_in[16];
  const float* dec_ffn_b2 = (const float*)d_in[17];
  const float* dec_ln     = (const float*)d_in[18];
  const float* dec_fln    = (const float*)d_in[19];

  float* out = (float*)d_out;
  float* ws  = (float*)d_ws;

  const int M     = TBB * TT;       // 1024 rows
  const size_t SB = (size_t)M * TD; // 524288 floats per activation buffer

  float* x   = ws;
  float* h   = x   + SB;
  float* q   = h   + SB;
  float* k   = q   + SB;
  float* v   = k   + SB;
  float* o   = v   + SB;
  float* mem = o   + SB;
  float* y   = mem + SB;
  float* ffn = y   + SB;            // 1024 * 2048

  const int total = M * TD;
  const dim3 agrid(THD, TBB);       // (head, batch)
  const size_t WSTRIDE = (size_t)TD * TD;   // 262144

  // ===================== encoder =====================
  embed_kernel<<<(total + 255) / 256, 256, 0, stream>>>(src, tok_emb, pos_emb, x, total);
  for (int i = 0; i < TNL; ++i) {
    const float* lnb = enc_ln + (size_t)i * 2 * 2 * TD;
    const float* w   = enc_attn_w + (size_t)i * 4 * WSTRIDE;
    const float* bb  = enc_attn_b + (size_t)i * 4 * TD;

    layernorm_kernel<<<M, 128, 0, stream>>>(x, lnb, lnb + TD, h);
    gemm(stream, h, w + 0 * WSTRIDE, bb + 0 * TD, nullptr, q, M, TD, TD, 0);
    gemm(stream, h, w + 1 * WSTRIDE, bb + 1 * TD, nullptr, k, M, TD, TD, 0);
    gemm(stream, h, w + 2 * WSTRIDE, bb + 2 * TD, nullptr, v, M, TD, TD, 0);
    attention_kernel<<<agrid, 128, 0, stream>>>(q, k, v, o, src, 0, TT, TS);
    gemm(stream, o, w + 3 * WSTRIDE, bb + 3 * TD, x, x, M, TD, TD, 0);

    layernorm_kernel<<<M, 128, 0, stream>>>(x, lnb + 2 * TD, lnb + 3 * TD, h);
    gemm(stream, h,   enc_ffn_w1 + (size_t)i * TD * TFF, enc_ffn_b1 + (size_t)i * TFF,
         nullptr, ffn, M, TFF, TD, GFLAG_RELU);
    gemm(stream, ffn, enc_ffn_w2 + (size_t)i * TFF * TD, enc_ffn_b2 + (size_t)i * TD,
         x, x, M, TD, TFF, 0);
  }
  layernorm_kernel<<<M, 128, 0, stream>>>(x, enc_fln, enc_fln + TD, mem);

  // ===================== decoder =====================
  embed_kernel<<<(total + 255) / 256, 256, 0, stream>>>(tgt, tok_emb, pos_emb, y, total);
  for (int i = 0; i < TNL; ++i) {
    const float* lnb = dec_ln + (size_t)i * 3 * 2 * TD;
    const float* w   = dec_attn_w + (size_t)i * 8 * WSTRIDE;
    const float* bb  = dec_attn_b + (size_t)i * 8 * TD;

    // masked self-attention
    layernorm_kernel<<<M, 128, 0, stream>>>(y, lnb, lnb + TD, h);
    gemm(stream, h, w + 0 * WSTRIDE, bb + 0 * TD, nullptr, q, M, TD, TD, 0);
    gemm(stream, h, w + 1 * WSTRIDE, bb + 1 * TD, nullptr, k, M, TD, TD, 0);
    gemm(stream, h, w + 2 * WSTRIDE, bb + 2 * TD, nullptr, v, M, TD, TD, 0);
    attention_kernel<<<agrid, 128, 0, stream>>>(q, k, v, o, tgt, 1, TT, TT);
    gemm(stream, o, w + 3 * WSTRIDE, bb + 3 * TD, y, y, M, TD, TD, 0);

    // cross-attention vs encoder memory
    layernorm_kernel<<<M, 128, 0, stream>>>(y, lnb + 2 * TD, lnb + 3 * TD, h);
    gemm(stream, h,   w + 4 * WSTRIDE, bb + 4 * TD, nullptr, q, M, TD, TD, 0);
    gemm(stream, mem, w + 5 * WSTRIDE, bb + 5 * TD, nullptr, k, M, TD, TD, 0);
    gemm(stream, mem, w + 6 * WSTRIDE, bb + 6 * TD, nullptr, v, M, TD, TD, 0);
    attention_kernel<<<agrid, 128, 0, stream>>>(q, k, v, o, src, 0, TT, TS);
    gemm(stream, o, w + 7 * WSTRIDE, bb + 7 * TD, y, y, M, TD, TD, 0);

    // FFN
    layernorm_kernel<<<M, 128, 0, stream>>>(y, lnb + 4 * TD, lnb + 5 * TD, h);
    gemm(stream, h,   dec_ffn_w1 + (size_t)i * TD * TFF, dec_ffn_b1 + (size_t)i * TFF,
         nullptr, ffn, M, TFF, TD, GFLAG_RELU);
    gemm(stream, ffn, dec_ffn_w2 + (size_t)i * TFF * TD, dec_ffn_b2 + (size_t)i * TD,
         y, y, M, TD, TFF, 0);
  }

  // ===================== generator =====================
  layernorm_kernel<<<M, 128, 0, stream>>>(y, dec_fln, dec_fln + TD, h);
  // logits = h @ tok_emb^T -> straight into d_out, then in-place log_softmax
  gemm(stream, h, tok_emb, nullptr, nullptr, out, M, TV, TD, GFLAG_BTRANS);
  log_softmax_kernel<<<M, 256, 0, stream>>>(out);
}
